// TransformerEncoderLayer_DAttention_15350213116198
// MI455X (gfx1250) — compile-verified, matrix operand order checked by reading, not run on hardware
//
#include <hip/hip_runtime.h>

typedef _Float16 h16;
typedef __attribute__((ext_vector_type(16))) _Float16 v16h;
typedef __attribute__((ext_vector_type(8)))  _Float16 v8h;
typedef __attribute__((ext_vector_type(8)))  float    v8f;

#define B_      64
#define C_      256
#define HH      20
#define WW      20
#define NPIX    400
#define CM_     2048
#define HEADS_  8
#define GROUPS_ 4
#define CG_     64
#define HC_     32
#define NPAD    416
#define SCALE_  0.17677669529663689f   // 32^-0.5

__device__ __forceinline__ float gelu_exact(float x) {
  return 0.5f * x * (1.0f + erff(x * 0.70710678118654752440f));
}
__device__ __forceinline__ float wred_sum(float v) {
  #pragma unroll
  for (int o = 16; o > 0; o >>= 1) v += __shfl_xor(v, o, 32);
  return v;
}
__device__ __forceinline__ float wred_max(float v) {
  #pragma unroll
  for (int o = 16; o > 0; o >>= 1) v = fmaxf(v, __shfl_xor(v, o, 32));
  return v;
}

union Frag { v16h v; v8h h[2]; };

// gfx1250 async global->LDS copy of 16 bytes (ASYNCcnt-tracked).
__device__ __forceinline__ void async_b128_to_lds(unsigned lds_byte_off,
                                                  const h16* gaddr) {
  asm volatile("global_load_async_to_lds_b128 %0, %1, off"
               :: "v"(lds_byte_off), "v"(gaddr) : "memory");
}
__device__ __forceinline__ void wait_asynccnt0() {
  asm volatile("s_wait_asynccnt 0x0" ::: "memory");
}

// ---------------------------------------------------------------------------
// WMMA GEMM with async-LDS staged activations:
//   out[b] = A(MxK) @ X_b(KxN) + bias, X stored (N,K) f16, K == 256.
// Block = 8 waves: one 16-col n-tile staged once into LDS, 8 m-tiles.
// grid.x = (M/16/8) * (N/16), grid.y = batch.
// Epilogues: f32 (b,M,N), f16 transposed (b,N,M), f16 natural padded.
// ---------------------------------------------------------------------------
__global__ void k_gemm(const h16* __restrict__ A, const h16* __restrict__ X,
                       const float* __restrict__ bias,
                       float* __restrict__ outF, h16* __restrict__ outT,
                       h16* __restrict__ outN,
                       int M, int N, int K, int natStride, int doGelu)
{
  __shared__ h16 sX[16 * C_];          // 8 KB: one (16 x K) activation tile
  const int wave = threadIdx.x >> 5;
  const int lane = threadIdx.x & 31;
  const int l15  = lane & 15;
  const int hb   = (lane >> 4) ? 8 : 0;     // half-select per 16-bit A/B layout
  const int tilesN = N >> 4;
  const int nt = blockIdx.x % tilesN;
  const int mg = blockIdx.x / tilesN;
  const int b  = blockIdx.y;
  const int n0 = nt << 4;
  const int m0 = (mg * 8 + wave) << 4;

  // stage X rows [n0, n0+16) x K halves into LDS via async b128 copies
  {
    const int kc = K >> 3;                       // 16B chunks per row
    const h16* gbase = X + ((size_t)b * N + n0) * K;
    for (int ch = threadIdx.x; ch < 16 * kc; ch += 256) {
      const int row = ch / kc;
      const int col = (ch % kc) << 3;
      async_b128_to_lds((unsigned)(size_t)&sX[row * K + col],
                        gbase + (size_t)row * K + col);
    }
    wait_asynccnt0();
    __syncthreads();
  }

  const h16* Arow = A + (size_t)(m0 + l15) * K + hb;
  const h16* Xl   = &sX[l15 * K + hb];

  v8f acc = {};
  for (int k0 = 0; k0 < K; k0 += 32) {
    __builtin_prefetch(Arow + k0 + 128, 0, 3);   // near-cache prefetch, next A
    Frag a, x;
    a.h[0] = *(const v8h*)(Arow + k0);
    a.h[1] = *(const v8h*)(Arow + k0 + 16);
    x.h[0] = *(const v8h*)(Xl + k0);
    x.h[1] = *(const v8h*)(Xl + k0 + 16);
    acc = __builtin_amdgcn_wmma_f32_16x16x32_f16(false, a.v, false, x.v,
                                                 (short)0, acc, false, false);
  }
  const int n = n0 + l15;
  #pragma unroll
  for (int r = 0; r < 8; ++r) {
    int m = m0 + r + hb;
    float v = acc[r] + bias[m];
    if (doGelu) v = gelu_exact(v);
    if (outF) outF[((size_t)b * M + m) * N + n] = v;
    if (outT) outT[((size_t)b * N + n) * M + m] = (h16)v;
    if (outN) outN[((size_t)b * M + m) * natStride + n] = (h16)v;
  }
}

// ---------------------------------------------------------------------------
// Fused attention: per (head, 16-row tile): S=qT@k (WMMA) -> LDS, scaled
// softmax in LDS, P@V^T (WMMA, K split over all 8 waves, LDS reduction).
// Output O^T f32 (b,n,C).
// ---------------------------------------------------------------------------
__global__ void k_attn(const h16* __restrict__ qT, const h16* __restrict__ kT,
                       const h16* __restrict__ vh, float* __restrict__ outT)
{
  __shared__ float sS[16][NPAD];   // scores, then partial-accumulator scratch
  __shared__ h16   sP[16][NPAD];

  const int wave = threadIdx.x >> 5;
  const int lane = threadIdx.x & 31;
  const int l15  = lane & 15;
  const int hb   = (lane >> 4) ? 8 : 0;
  const int bh   = blockIdx.y;
  const int b    = bh >> 3;
  const int headOff = (bh & 7) << 5;
  const int m0   = blockIdx.x << 4;

  // ---- stage A: S tiles (M=16 rows, 25 N-tiles, K=32) ----
  const h16* Qrow = qT + ((size_t)b * NPIX + (m0 + l15)) * C_ + headOff + hb;
  Frag a;
  a.h[0] = *(const v8h*)(Qrow);
  a.h[1] = *(const v8h*)(Qrow + 16);
  for (int t = wave; t < 25; t += 8) {
    const int n0 = t << 4;
    const h16* Krow = kT + ((size_t)b * NPIX + (n0 + l15)) * C_ + headOff + hb;
    Frag x;
    x.h[0] = *(const v8h*)(Krow);
    x.h[1] = *(const v8h*)(Krow + 16);
    v8f acc = {};
    acc = __builtin_amdgcn_wmma_f32_16x16x32_f16(false, a.v, false, x.v,
                                                 (short)0, acc, false, false);
    #pragma unroll
    for (int r = 0; r < 8; ++r) sS[r + hb][n0 + l15] = acc[r];
  }
  __syncthreads();

  // ---- stage B: softmax over each of 16 rows (2 rows per wave) ----
  #pragma unroll
  for (int rr = 0; rr < 2; ++rr) {
    const int row = wave * 2 + rr;
    float mx = -3.4e38f;
    for (int j = lane; j < NPIX; j += 32) {
      float v = sS[row][j] * SCALE_;
      sS[row][j] = v;
      mx = fmaxf(mx, v);
    }
    mx = wred_max(mx);
    float sum = 0.f;
    for (int j = lane; j < NPIX; j += 32) {
      float e = __expf(sS[row][j] - mx);
      sS[row][j] = e;
      sum += e;
    }
    sum = wred_sum(sum);
    const float inv = 1.0f / sum;
    for (int j = lane; j < NPIX; j += 32) sP[row][j] = (h16)(sS[row][j] * inv);
    if (lane < 16) sP[row][NPIX + lane] = (h16)0.0f;
  }
  __syncthreads();

  // ---- stage C: O^T = P(16x416) @ V^T(416x32) ----
  // All 8 waves active: wave = kslice*2 + ctile; partials reduced via LDS.
  float* part = &sS[0][0];               // sS is dead; reuse as 8x256 scratch
  {
    const int c0 = (wave & 1) << 4;      // channel tile 0 / 16
    const int ks = wave >> 1;            // K slice 0..3 over 13 k-steps
    const int kbeg = ks * 4;
    const int kend = (kbeg + 4 < 13) ? (kbeg + 4) : 13;
    const h16* Vrow = vh + ((size_t)b * C_ + headOff + c0 + l15) * NPAD + hb;
    v8f acc = {};
    for (int kt = kbeg; kt < kend; ++kt) {
      const int kk = kt * 32;
      Frag p, x;
      const h16* Pp = &sP[l15][kk + hb];
      p.h[0] = *(const v8h*)(Pp);
      p.h[1] = *(const v8h*)(Pp + 16);
      x.h[0] = *(const v8h*)(Vrow + kk);
      x.h[1] = *(const v8h*)(Vrow + kk + 16);
      acc = __builtin_amdgcn_wmma_f32_16x16x32_f16(false, p.v, false, x.v,
                                                   (short)0, acc, false, false);
    }
    #pragma unroll
    for (int r = 0; r < 8; ++r)
      part[wave * 256 + (r + hb) * 16 + l15] = acc[r];
  }
  __syncthreads();

  if (wave < 2) {                         // wave == ctile index
    const int c0 = wave << 4;
    #pragma unroll
    for (int r = 0; r < 8; ++r) {
      const int rl = r + hb;
      float v = part[(0 * 2 + wave) * 256 + rl * 16 + l15]
              + part[(1 * 2 + wave) * 256 + rl * 16 + l15]
              + part[(2 * 2 + wave) * 256 + rl * 16 + l15]
              + part[(3 * 2 + wave) * 256 + rl * 16 + l15];
      outT[((size_t)b * NPIX + m0 + rl) * C_ + headOff + c0 + l15] = v;
    }
  }
}

// ---------------------------------------------------------------------------
// Fused FFN: async-staged s^T tile; hidden 16x2048 GELU tile in LDS (64KB);
// then fc2 GEMM with B-frags from LDS.
// ---------------------------------------------------------------------------
__global__ void k_ffn(const h16* __restrict__ sT, const h16* __restrict__ w1,
                      const float* __restrict__ b1, const h16* __restrict__ w2,
                      const float* __restrict__ b2, float* __restrict__ ffn)
{
  __shared__ h16 sXf[16 * C_];  // staged input tile (8 KB)
  __shared__ h16 sH[16][CM_];   // hidden transposed: [pixel][cm] (64 KB)
  const int wave = threadIdx.x >> 5;
  const int lane = threadIdx.x & 31;
  const int l15  = lane & 15;
  const int hb   = (lane >> 4) ? 8 : 0;
  const int b    = blockIdx.y;
  const int n0   = blockIdx.x << 4;

  // async stage s^T rows [n0, n0+16) x 256 into LDS
  {
    const h16* gbase = sT + ((size_t)b * NPIX + n0) * C_;
    for (int ch = threadIdx.x; ch < 16 * (C_ >> 3); ch += 256) {
      const int row = ch >> 5;                 // C_/8 == 32 chunks per row
      const int col = (ch & 31) << 3;
      async_b128_to_lds((unsigned)(size_t)&sXf[row * C_ + col],
                        gbase + (size_t)row * C_ + col);
    }
    wait_asynccnt0();
    __syncthreads();
  }
  const h16* Xl = &sXf[l15 * C_ + hb];

  // ---- stage 1: h = gelu(fc1 @ s + b1), 128 M-tiles over 8 waves ----
  for (int mt = wave; mt < CM_ / 16; mt += 8) {
    const h16* Arow = w1 + (size_t)(mt * 16 + l15) * C_ + hb;
    v8f acc = {};
    for (int k0 = 0; k0 < C_; k0 += 32) {
      Frag a, xx;
      a.h[0]  = *(const v8h*)(Arow + k0);
      a.h[1]  = *(const v8h*)(Arow + k0 + 16);
      xx.h[0] = *(const v8h*)(Xl + k0);
      xx.h[1] = *(const v8h*)(Xl + k0 + 16);
      acc = __builtin_amdgcn_wmma_f32_16x16x32_f16(false, a.v, false, xx.v,
                                                   (short)0, acc, false, false);
    }
    #pragma unroll
    for (int r = 0; r < 8; ++r) {
      int cm = mt * 16 + r + hb;
      sH[l15][cm] = (h16)gelu_exact(acc[r] + b1[cm]);
    }
  }
  __syncthreads();

  // ---- stage 2: ffn = fc2 @ h + b2, 16 M-tiles over 8 waves ----
  for (int mt = wave; mt < C_ / 16; mt += 8) {
    const h16* Arow = w2 + (size_t)(mt * 16 + l15) * CM_ + hb;
    const h16* Hrow = &sH[l15][hb];
    v8f acc = {};
    for (int k0 = 0; k0 < CM_; k0 += 32) {
      __builtin_prefetch(Arow + k0 + 128, 0, 3);
      Frag a, xx;
      a.h[0]  = *(const v8h*)(Arow + k0);
      a.h[1]  = *(const v8h*)(Arow + k0 + 16);
      xx.h[0] = *(const v8h*)(Hrow + k0);
      xx.h[1] = *(const v8h*)(Hrow + k0 + 16);
      acc = __builtin_amdgcn_wmma_f32_16x16x32_f16(false, a.v, false, xx.v,
                                                   (short)0, acc, false, false);
    }
    #pragma unroll
    for (int r = 0; r < 8; ++r) {
      int ch = mt * 16 + r + hb;
      ffn[((size_t)b * C_ + ch) * NPIX + n0 + l15] = acc[r] + b2[ch];
    }
  }
}

// ---------------------------------------------------------------------------
// Elementwise / normalization / conv helper kernels (f32 VALU)
// ---------------------------------------------------------------------------
__global__ void k_cvt(const float* __restrict__ in, h16* __restrict__ out, int n) {
  int t = blockIdx.x * 256 + threadIdx.x;
  if (t < n) out[t] = (h16)in[t];
}
__global__ void k_zeroh(h16* __restrict__ out, int n) {
  int t = blockIdx.x * 256 + threadIdx.x;
  if (t < n) out[t] = (h16)0.0f;
}
// (b,C,n) f32 -> (b,n,C) f16
__global__ void k_transpose(const float* __restrict__ in, h16* __restrict__ out) {
  int t = blockIdx.x * 256 + threadIdx.x;
  if (t >= B_ * NPIX * C_) return;
  int c = t % C_, p = (t / C_) % NPIX, b = t / (C_ * NPIX);
  out[t] = (h16)in[((size_t)b * C_ + c) * NPIX + p];
}
// depthwise 3x3, pad 1, per-channel weights (CC,3,3)
__global__ void k_dw3x3(const float* __restrict__ in, const float* __restrict__ w,
                        const float* __restrict__ bias, float* __restrict__ out,
                        int NB, int CC)
{
  int t = blockIdx.x * 256 + threadIdx.x;
  if (t >= NB * CC * NPIX) return;
  int x = t % WW, y = (t / WW) % HH, c = (t / NPIX) % CC;
  int nb = t / (NPIX * CC);
  const float* ip = in + ((size_t)nb * CC + c) * NPIX;
  const float* wp = w + c * 9;
  float acc = bias[c];
  #pragma unroll
  for (int dy = 0; dy < 3; ++dy) {
    int yy = y + dy - 1;
    if (yy < 0 || yy >= HH) continue;
    #pragma unroll
    for (int dx = 0; dx < 3; ++dx) {
      int xx = x + dx - 1;
      if (xx < 0 || xx >= WW) continue;
      acc += ip[yy * WW + xx] * wp[dy * 3 + dx];
    }
  }
  out[t] = acc;
}
// offset head: LN(64,eps1e-5)+GELU+pointwise(2x64)+tanh*range+ref -> pos(y,x)
__global__ void k_offhead(const float* __restrict__ off1, const float* __restrict__ lnw,
                          const float* __restrict__ lnb, const float* __restrict__ pw,
                          float* __restrict__ pos)
{
  int t = blockIdx.x * 256 + threadIdx.x;
  if (t >= B_ * GROUPS_ * NPIX) return;
  int bg = t / NPIX, p = t % NPIX;
  float v[CG_];
  float s = 0.f;
  #pragma unroll 8
  for (int c = 0; c < CG_; ++c) { v[c] = off1[((size_t)bg * CG_ + c) * NPIX + p]; s += v[c]; }
  float mean = s * (1.0f / CG_);
  float q = 0.f;
  #pragma unroll 8
  for (int c = 0; c < CG_; ++c) { float d = v[c] - mean; q += d * d; }
  float rstd = rsqrtf(q * (1.0f / CG_) + 1e-5f);
  float py = 0.f, px = 0.f;
  #pragma unroll 8
  for (int c = 0; c < CG_; ++c) {
    float u = gelu_exact(lnw[c] * (v[c] - mean) * rstd + lnb[c]);
    py += pw[c] * u;
    px += pw[CG_ + c] * u;
  }
  const float rng = 4.0f / 19.0f;
  int i = p / WW, j = p % WW;
  pos[(size_t)t * 2 + 0] = tanhf(py) * rng + ((0.5f + i) / 19.0f * 2.0f - 1.0f);
  pos[(size_t)t * 2 + 1] = tanhf(px) * rng + ((0.5f + j) / 19.0f * 2.0f - 1.0f);
}
// bilinear grid sample (zeros padding, align_corners=True) -> x_s^T f16 (b,n,C)
__global__ void k_sample(const float* __restrict__ src, const float* __restrict__ pos,
                         h16* __restrict__ xsT)
{
  int t = blockIdx.x * 256 + threadIdx.x;
  if (t >= B_ * NPIX * C_) return;
  int c = t % C_, p = (t / C_) % NPIX, b = t / (C_ * NPIX);
  int g = c >> 6;
  const float* pp = pos + (((size_t)(b * GROUPS_ + g)) * NPIX + p) * 2;
  float y = (pp[0] + 1.0f) * 0.5f * (HH - 1);
  float x = (pp[1] + 1.0f) * 0.5f * (WW - 1);
  float x0f = floorf(x), y0f = floorf(y);
  int x0 = (int)x0f, y0 = (int)y0f;
  float fx = x - x0f, fy = y - y0f;
  const float* im = src + ((size_t)b * C_ + c) * NPIX;
  auto F = [&](int yy, int xx) -> float {
    return (xx >= 0 && xx < WW && yy >= 0 && yy < HH) ? im[yy * WW + xx] : 0.0f;
  };
  float val = F(y0, x0) * (1.f - fx) * (1.f - fy) + F(y0 + 1, x0) * (1.f - fx) * fy
            + F(y0, x0 + 1) * fx * (1.f - fy)     + F(y0 + 1, x0 + 1) * fx * fy;
  xsT[t] = (h16)val;
}
// oinT[b,n,c] = f16(attnOutT[b,n,c] + lepe[b,c,n])
__global__ void k_combine(const float* __restrict__ aT, const float* __restrict__ lepe,
                          h16* __restrict__ oinT)
{
  int t = blockIdx.x * 256 + threadIdx.x;
  if (t >= B_ * NPIX * C_) return;
  int c = t % C_, p = (t / C_) % NPIX, b = t / (C_ * NPIX);
  oinT[t] = (h16)(aT[t] + lepe[((size_t)b * C_ + c) * NPIX + p]);
}
// LayerNorm over C=256 of (X+Y); wave per pixel; writes f32 (b,C,n) + f16 (b,n,C)
__global__ void k_ln(const float* __restrict__ X, const float* __restrict__ Y,
                     const float* __restrict__ w, const float* __restrict__ bb,
                     float eps, float* __restrict__ outF, h16* __restrict__ outT)
{
  const int wave = threadIdx.x >> 5, lane = threadIdx.x & 31;
  const int pix = blockIdx.x * 8 + wave;
  if (pix >= B_ * NPIX) return;
  const int b = pix / NPIX, p = pix % NPIX;
  float v[8];
  float s = 0.f;
  #pragma unroll
  for (int i = 0; i < 8; ++i) {
    int c = lane + i * 32;
    size_t idx = ((size_t)b * C_ + c) * NPIX + p;
    v[i] = X[idx] + Y[idx];
    s += v[i];
  }
  s = wred_sum(s);
  float mean = s * (1.0f / C_);
  float q = 0.f;
  #pragma unroll
  for (int i = 0; i < 8; ++i) { float d = v[i] - mean; q += d * d; }
  q = wred_sum(q);
  float rstd = rsqrtf(q * (1.0f / C_) + eps);
  #pragma unroll
  for (int i = 0; i < 8; ++i) {
    int c = lane + i * 32;
    float o = w[c] * (v[i] - mean) * rstd + bb[c];
    if (outF) outF[((size_t)b * C_ + c) * NPIX + p] = o;
    if (outT) outT[((size_t)b * NPIX + p) * C_ + c] = (h16)o;
  }
}

// ---------------------------------------------------------------------------
extern "C" void kernel_launch(void* const* d_in, const int* in_sizes, int n_in,
                              void* d_out, int out_size, void* d_ws, size_t ws_size,
                              hipStream_t stream) {
  (void)in_sizes; (void)n_in; (void)out_size; (void)ws_size;
  const float* src      = (const float*)d_in[0];
  const float* w_q      = (const float*)d_in[1];
  const float* b_q      = (const float*)d_in[2];
  const float* w_k      = (const float*)d_in[3];
  const float* b_k      = (const float*)d_in[4];
  const float* w_v      = (const float*)d_in[5];
  const float* b_v      = (const float*)d_in[6];
  const float* w_o      = (const float*)d_in[7];
  const float* b_o      = (const float*)d_in[8];
  const float* off_dw_w = (const float*)d_in[9];
  const float* off_dw_b = (const float*)d_in[10];
  const float* off_ln_w = (const float*)d_in[11];
  const float* off_ln_b = (const float*)d_in[12];
  const float* off_pw_w = (const float*)d_in[13];
  const float* rpe_w    = (const float*)d_in[14];
  const float* rpe_b    = (const float*)d_in[15];
  const float* fc1_w    = (const float*)d_in[16];
  const float* fc1_b    = (const float*)d_in[17];
  const float* fc2_w    = (const float*)d_in[18];
  const float* fc2_b    = (const float*)d_in[19];
  const float* n1_w     = (const float*)d_in[20];
  const float* n1_b     = (const float*)d_in[21];
  const float* n2_w     = (const float*)d_in[22];
  const float* n2_b     = (const float*)d_in[23];
  float* out = (float*)d_out;

  char* ws = (char*)d_ws;
  size_t off = 0;
  auto alloc = [&](size_t bytes) -> void* {
    off = (off + 255) & ~(size_t)255;
    void* p = ws + off;
    off += bytes;
    return p;
  };
  const size_t NE  = (size_t)B_ * C_ * NPIX;       // 6.55M elems
  h16*   wqh  = (h16*)alloc((size_t)C_ * C_ * 2);
  h16*   wkh  = (h16*)alloc((size_t)C_ * C_ * 2);
  h16*   wvh  = (h16*)alloc((size_t)C_ * C_ * 2);
  h16*   woh  = (h16*)alloc((size_t)C_ * C_ * 2);
  h16*   w1h  = (h16*)alloc((size_t)CM_ * C_ * 2);
  h16*   w2h  = (h16*)alloc((size_t)C_ * CM_ * 2);
  h16*   abuf = (h16*)alloc(NE * 2);               // srcT -> x_sT
  float* bbuf = (float*)alloc(NE * 4);             // q f32 -> s f32
  h16*   qT   = (h16*)alloc(NE * 2);
  float* dbuf = (float*)alloc(NE * 4);             // off1 -> attnOutT -> y
  float* pos  = (float*)alloc((size_t)B_ * GROUPS_ * NPIX * 2 * 4);
  h16*   kT   = (h16*)alloc(NE * 2);
  h16*   vh   = (h16*)alloc((size_t)B_ * C_ * NPAD * 2);
  float* hbuf = (float*)alloc(NE * 4);             // lepe -> ffn
  h16*   ibuf = (h16*)alloc(NE * 2);               // oinT -> sT

  const int TPB = 256;
  const int gElem = (int)((NE + TPB - 1) / TPB);          // 25600
  const dim3 gGemm(50, B_);              // (16 mtiles / 8) * 25 ntiles = 50

  // 1. weights -> f16
  k_cvt<<<(C_*C_+TPB-1)/TPB, TPB, 0, stream>>>(w_q, wqh, C_*C_);
  k_cvt<<<(C_*C_+TPB-1)/TPB, TPB, 0, stream>>>(w_k, wkh, C_*C_);
  k_cvt<<<(C_*C_+TPB-1)/TPB, TPB, 0, stream>>>(w_v, wvh, C_*C_);
  k_cvt<<<(C_*C_+TPB-1)/TPB, TPB, 0, stream>>>(w_o, woh, C_*C_);
  k_cvt<<<(CM_*C_+TPB-1)/TPB, TPB, 0, stream>>>(fc1_w, w1h, CM_*C_);
  k_cvt<<<(CM_*C_+TPB-1)/TPB, TPB, 0, stream>>>(fc2_w, w2h, CM_*C_);

  // 2. srcT f16
  k_transpose<<<gElem, TPB, 0, stream>>>(src, abuf);
  // 3. q = Wq @ src  (f32 + transposed f16 for attention A-frags)
  k_gemm<<<gGemm, TPB, 0, stream>>>(wqh, abuf, b_q, bbuf, qT, nullptr,
                                    C_, NPIX, C_, 0, 0);
  // 4. offset depthwise conv (view q as (B*G, CG, H, W))
  k_dw3x3<<<gElem, TPB, 0, stream>>>(bbuf, off_dw_w, off_dw_b, dbuf,
                                     B_ * GROUPS_, CG_);
  // 5. LN + GELU + pointwise + tanh + ref -> pos
  k_offhead<<<(B_*GROUPS_*NPIX + TPB - 1)/TPB, TPB, 0, stream>>>(
      dbuf, off_ln_w, off_ln_b, off_pw_w, pos);
  // 6. grid sample -> x_s^T f16
  k_sample<<<gElem, TPB, 0, stream>>>(src, pos, abuf);
  // 7/8. k,v projections (zero-pad v to NPAD first)
  {
    int nvh = B_ * C_ * NPAD;
    k_zeroh<<<(nvh + TPB - 1)/TPB, TPB, 0, stream>>>(vh, nvh);
  }
  k_gemm<<<gGemm, TPB, 0, stream>>>(wkh, abuf, b_k, nullptr, kT, nullptr,
                                    C_, NPIX, C_, 0, 0);
  k_gemm<<<gGemm, TPB, 0, stream>>>(wvh, abuf, b_v, nullptr, nullptr, vh,
                                    C_, NPIX, C_, NPAD, 0);
  // 9. fused attention -> O^T f32 (b,n,C)
  k_attn<<<dim3(NPIX/16, B_*HEADS_), TPB, 0, stream>>>(qT, kT, vh, dbuf);
  // 10. lepe depthwise conv on q
  k_dw3x3<<<gElem, TPB, 0, stream>>>(bbuf, rpe_w, rpe_b, hbuf, B_, C_);
  // 11. combine O^T + lepe -> f16 (b,n,C)
  k_combine<<<gElem, TPB, 0, stream>>>(dbuf, hbuf, ibuf);
  // 12. output projection y = Wo @ (out)  (f32 (b,C,n), reuses dbuf)
  k_gemm<<<gGemm, TPB, 0, stream>>>(woh, ibuf, b_o, dbuf, nullptr, nullptr,
                                    C_, NPIX, C_, 0, 0);
  // 13. s = LN(src + y)  -> f32 (bbuf) + f16 transposed (ibuf)
  k_ln<<<(B_*NPIX)/8, TPB, 0, stream>>>(src, dbuf, n1_w, n1_b, 1e-6f, bbuf, ibuf);
  // 14. fused FFN -> ffn f32 (hbuf)
  k_ffn<<<dim3(NPIX/16, B_), TPB, 0, stream>>>(ibuf, w1h, fc1_b, w2h, fc2_b, hbuf);
  // 15. out = LN(s + ffn)
  k_ln<<<(B_*NPIX)/8, TPB, 0, stream>>>(bbuf, hbuf, n2_w, n2_b, 1e-6f, out, nullptr);
}
